// CRFTModule_74036646248836
// MI455X (gfx1250) — compile-verified, matrix operand order checked by reading, not run on hardware
//
#include <hip/hip_runtime.h>
#include <math.h>

// ---------------------------------------------------------------------------
// CRFT fused module for gfx1250 (MI455X), bf16 WMMA path.
// B=8, S=4096, H=1024, A_DIM=8, N_ADAPT=4. Rows = B*S = 32768.
// 64 rows per block -> 4x B-operand reuse per load, 0.5 GB L2 weight traffic.
// ---------------------------------------------------------------------------

typedef __attribute__((ext_vector_type(16))) __bf16 v16bf;
typedef __attribute__((ext_vector_type(8)))  float  v8f;

#define NROWS   32768
#define H_DIM   1024
#define K_HALF  512
#define MB      64                    // rows per block (4 M-tiles)
#define LOGIT07 0.8472978603872037f   // ln(0.7/0.3): sigmoid(z)>0.7 <=> z>this
#define SCALE_F 0.3f

// smem: x (MB*1024 bf16) + g (MB*32 bf16) + imp[MB] + mask[MB] + aw[4]
#define SMEM_X_BYTES   (MB * H_DIM * 2)            // 131072
#define SMEM_G_OFF     SMEM_X_BYTES
#define SMEM_G_BYTES   (MB * 32 * 2)               // 4096
#define SMEM_IMP_OFF   (SMEM_G_OFF + SMEM_G_BYTES)
#define SMEM_TOTAL     (SMEM_IMP_OFF + MB * 4 + MB * 4 + 16)

static __device__ __forceinline__ unsigned short f2bf(float f) {
    unsigned int u = __float_as_uint(f);
    u += 0x7FFFu + ((u >> 16) & 1u);          // round-to-nearest-even
    return (unsigned short)(u >> 16);
}
static __device__ __forceinline__ float gelu_exact(float x) {
    return 0.5f * x * (1.0f + erff(x * 0.70710678118654752f));
}
// K index inside a 32-deep bf16 WMMA operand tile for (lane, j) slot.
static __device__ __forceinline__ int kk_of(int lane, int j) {
    return ((j >> 3) & 1) * 16 + ((j >> 1) & 3) * 2 + ((lane >> 4) & 1) * 8 + (j & 1);
}

// ---------------- weight packing kernels (bf16, WMMA B-operand layout) -----
// Packed tile = 512 bf16; per-lane 16 contiguous bf16 (two b128 loads).

__global__ void crft_pack_w1(const float* __restrict__ w1, unsigned short* __restrict__ w1p) {
    int o = blockIdx.x * 256 + threadIdx.x;              // 32*32 tiles * 512
    int tile = o >> 9, within = o & 511;
    int nt = tile >> 5, kt = tile & 31;
    int lane = within >> 4, j = within & 15;
    int k = kt * 32 + kk_of(lane, j);
    int n = nt * 16 + (lane & 15);
    w1p[o] = f2bf(w1[k * K_HALF + n]);
}

__global__ void crft_pack_dw(const float* __restrict__ dw, unsigned short* __restrict__ dwp) {
    int o = blockIdx.x * 256 + threadIdx.x;              // 2*32 tiles * 512
    int tile = o >> 9, within = o & 511;
    int nt = tile >> 5, kt = tile & 31;
    int lane = within >> 4, j = within & 15;
    int k = kt * 32 + kk_of(lane, j);
    int c = nt * 16 + (lane & 15);                       // c = a*8 + d
    dwp[o] = f2bf(dw[(c >> 3) * 8192 + k * 8 + (c & 7)]);
}

__global__ void crft_pack_up(const float* __restrict__ uw, unsigned short* __restrict__ upp) {
    int o = blockIdx.x * 256 + threadIdx.x;              // 64 tiles * 512 (K=32)
    int nt = o >> 9, within = o & 511;
    int lane = within >> 4, j = within & 15;
    int c = kk_of(lane, j);                              // K row = a*8 + d
    int h = nt * 16 + (lane & 15);
    upp[o] = f2bf(uw[(c >> 3) * 8192 + (c & 7) * H_DIM + h]);
}

// ---------------- adapter weights (softmax) + weighted up-bias -------------

__global__ void crft_adapter(const float* __restrict__ x, const float* __restrict__ sw,
                             const float* __restrict__ sb, const float* __restrict__ ub,
                             float* __restrict__ awbuf, float* __restrict__ wbbuf) {
    __shared__ float red[256][4];
    __shared__ float aw_l[4];
    int b = blockIdx.x, t = threadIdx.x;
    const float* last = x + ((size_t)b * 4096 + 4095) * H_DIM;
    float p[4] = {0.f, 0.f, 0.f, 0.f};
    for (int k = t; k < H_DIM; k += 256) {
        float xv = last[k];
        #pragma unroll
        for (int a = 0; a < 4; ++a) p[a] += xv * sw[k * 4 + a];
    }
    #pragma unroll
    for (int a = 0; a < 4; ++a) red[t][a] = p[a];
    __syncthreads();
    for (int s = 128; s > 0; s >>= 1) {
        if (t < s)
            #pragma unroll
            for (int a = 0; a < 4; ++a) red[t][a] += red[t + s][a];
        __syncthreads();
    }
    if (t == 0) {
        float l[4], mx = -1e30f, sum = 0.f;
        #pragma unroll
        for (int a = 0; a < 4; ++a) { l[a] = red[0][a] + sb[a]; mx = fmaxf(mx, l[a]); }
        #pragma unroll
        for (int a = 0; a < 4; ++a) { l[a] = __expf(l[a] - mx); sum += l[a]; }
        #pragma unroll
        for (int a = 0; a < 4; ++a) { float v = l[a] / sum; aw_l[a] = v; awbuf[b * 4 + a] = v; }
    }
    __syncthreads();
    for (int h = t; h < H_DIM; h += 256) {
        float s = 0.f;
        #pragma unroll
        for (int a = 0; a < 4; ++a) s += aw_l[a] * ub[a * H_DIM + h];
        wbbuf[b * H_DIM + h] = s;
    }
}

// ---------------- main fused kernel: 64 rows per block ---------------------

__global__ __launch_bounds__(256) void crft_main(
    const float* __restrict__ x, const float* __restrict__ b1,
    const float* __restrict__ w2, const float* __restrict__ b2,
    const float* __restrict__ db,
    const unsigned short* __restrict__ w1p,
    const unsigned short* __restrict__ dwp,
    const unsigned short* __restrict__ upp,
    const float* __restrict__ awbuf, const float* __restrict__ wbbuf,
    float* __restrict__ out)
{
    extern __shared__ char smem[];
    unsigned short* x_lds = (unsigned short*)smem;                 // [mt(4)][kt(32)] tiles*512
    unsigned short* g_lds = (unsigned short*)(smem + SMEM_G_OFF);  // [mt(4)] tiles*512
    float* imp   = (float*)(smem + SMEM_IMP_OFF);                  // MB
    float* maskv = imp + MB;                                       // MB
    float* aw_s  = maskv + MB;                                     // 4

    const int t    = threadIdx.x;
    const int lane = t & 31;
    const int w    = t >> 5;                 // wave id 0..7
    const int rowbase = blockIdx.x * MB;
    const int batch   = rowbase >> 12;       // / 4096 (MB divides 4096)

    if (t < MB) imp[t] = 0.f;
    if (t >= MB && t < MB + 4) aw_s[t - MB] = awbuf[batch * 4 + (t - MB)];

    // ---- stage x (64 x 1024 f32) into LDS as bf16, A-operand layout ----
    {
        int r  = t >> 2;                     // row 0..63
        int k0 = (t & 3) * 256;              // 256 K-elements per thread
        int mt = r >> 4, rr = r & 15;
        const float* xr = x + ((size_t)(rowbase + r)) * H_DIM + k0;
        for (int q = 0; q < 256; q += 4) {
            float4 v = *(const float4*)(xr + q);
            int kb  = k0 + q;
            int kt  = kb >> 5;
            int kk0 = kb & 31;               // even
            unsigned int p0 = (unsigned int)f2bf(v.x) | ((unsigned int)f2bf(v.y) << 16);
            unsigned int p1 = (unsigned int)f2bf(v.z) | ((unsigned int)f2bf(v.w) << 16);
            int base = (mt * 32 + kt) * 512;
            int l0 = rr + ((kk0 >> 3) & 1) * 16;
            int j0 = ((((kk0 >> 4) & 1) * 4) + ((kk0 >> 1) & 3)) * 2;
            *(unsigned int*)(x_lds + base + l0 * 16 + j0) = p0;
            int kk1 = kk0 + 2;
            int l1 = rr + ((kk1 >> 3) & 1) * 16;
            int j1 = ((((kk1 >> 4) & 1) * 4) + ((kk1 >> 1) & 3)) * 2;
            *(unsigned int*)(x_lds + base + l1 * 16 + j1) = p1;
        }
    }
    __syncthreads();

    const v8f zero = {};

    // ---- GEMM1: h = gelu(x @ pd_w1 + b1); importance += h . pd_w2 ----
    // 2 passes; per pass wave w owns N-tiles {w+16p, w+8+16p}, all 4 M-tiles.
    // Each B-tile register load feeds 4 WMMAs (one per M-tile).
    float rr_imp[4][8];
    #pragma unroll
    for (int mt = 0; mt < 4; ++mt)
        #pragma unroll
        for (int v = 0; v < 8; ++v) rr_imp[mt][v] = 0.f;

    for (int pass = 0; pass < 2; ++pass) {
        v8f acc[4][2];
        #pragma unroll
        for (int mt = 0; mt < 4; ++mt) { acc[mt][0] = zero; acc[mt][1] = zero; }
        for (int kt = 0; kt < 32; ++kt) {
            v16bf a0 = *((const v16bf*)(x_lds + (0 * 32 + kt) * 512 + lane * 16));
            v16bf a1 = *((const v16bf*)(x_lds + (1 * 32 + kt) * 512 + lane * 16));
            v16bf a2 = *((const v16bf*)(x_lds + (2 * 32 + kt) * 512 + lane * 16));
            v16bf a3 = *((const v16bf*)(x_lds + (3 * 32 + kt) * 512 + lane * 16));
            if (kt + 1 < 32) {               // gfx1250 global_prefetch_b8 on B stream
                #pragma unroll
                for (int i = 0; i < 2; ++i) {
                    int ntn = w + 8 * i + 16 * pass;
                    __builtin_prefetch(w1p + ((size_t)(ntn * 32 + kt + 1)) * 512 + lane * 16, 0, 3);
                }
            }
            #pragma unroll
            for (int i = 0; i < 2; ++i) {
                int nt = w + 8 * i + 16 * pass;
                v16bf bm = *((const v16bf*)(w1p + ((size_t)(nt * 32 + kt)) * 512 + lane * 16));
                acc[0][i] = __builtin_amdgcn_wmma_f32_16x16x32_bf16(false, a0, false, bm, (short)0, acc[0][i], false, false);
                acc[1][i] = __builtin_amdgcn_wmma_f32_16x16x32_bf16(false, a1, false, bm, (short)0, acc[1][i], false, false);
                acc[2][i] = __builtin_amdgcn_wmma_f32_16x16x32_bf16(false, a2, false, bm, (short)0, acc[2][i], false, false);
                acc[3][i] = __builtin_amdgcn_wmma_f32_16x16x32_bf16(false, a3, false, bm, (short)0, acc[3][i], false, false);
            }
        }
        #pragma unroll
        for (int i = 0; i < 2; ++i) {
            int n = (w + 8 * i + 16 * pass) * 16 + (lane & 15);
            float bb = b1[n], ww = w2[n];
            #pragma unroll
            for (int mt = 0; mt < 4; ++mt)
                #pragma unroll
                for (int v = 0; v < 8; ++v) {
                    float hv = gelu_exact(acc[mt][i][v] + bb);
                    rr_imp[mt][v] += hv * ww;
                }
        }
    }
    // reduce across the 16 lanes sharing each row group
    #pragma unroll
    for (int m = 1; m <= 8; m <<= 1)
        #pragma unroll
        for (int mt = 0; mt < 4; ++mt)
            #pragma unroll
            for (int v = 0; v < 8; ++v) rr_imp[mt][v] += __shfl_xor(rr_imp[mt][v], m, 32);
    if ((lane & 15) == 0) {
        int rb = (lane >> 4) * 8;            // lane0 -> rows 0..7, lane16 -> 8..15
        #pragma unroll
        for (int mt = 0; mt < 4; ++mt)
            #pragma unroll
            for (int v = 0; v < 8; ++v) atomicAdd(&imp[mt * 16 + rb + v], rr_imp[mt][v]);
    }
    __syncthreads();
    if (t < MB) {
        float z = imp[t] + b2[0];
        maskv[t] = (z > LOGIT07) ? 1.f : 0.f;   // sigmoid threshold moved pre-sigmoid
    }

    // ---- GEMM2: down (64 x 1024 x 32), gelu, scale by adapter weight ----
    {
        int mt  = w >> 1;                    // wave -> (M-tile, N-tile of 2)
        int nt2 = w & 1;
        v8f acc2 = zero;
        for (int kt = 0; kt < 32; ++kt) {
            v16bf a = *((const v16bf*)(x_lds + (mt * 32 + kt) * 512 + lane * 16));
            v16bf bm = *((const v16bf*)(dwp + ((size_t)(nt2 * 32 + kt)) * 512 + lane * 16));
            acc2 = __builtin_amdgcn_wmma_f32_16x16x32_bf16(false, a, false, bm, (short)0, acc2, false, false);
        }
        int c = nt2 * 16 + (lane & 15);      // c = a*8 + d in 0..31
        float awv = aw_s[c >> 3];
        float dbn = db[(c >> 3) * 8 + (c & 7)];
        #pragma unroll
        for (int v = 0; v < 8; ++v) {
            float g = gelu_exact(acc2[v] + dbn) * awv;
            int rl = v + (lane >> 4) * 8;    // row within M-tile, 0..15
            int l2 = rl + ((c >> 3) & 1) * 16;
            int j2 = ((((c >> 4) & 1) * 4) + ((c >> 1) & 3)) * 2 + (c & 1);
            g_lds[mt * 512 + l2 * 16 + j2] = f2bf(g);
        }
    }
    __syncthreads();

    // ---- GEMM3: weighted = g @ U (K=32, single WMMA per tile) + epilogue --
    v16bf ag0 = *((const v16bf*)(g_lds + 0 * 512 + lane * 16));
    v16bf ag1 = *((const v16bf*)(g_lds + 1 * 512 + lane * 16));
    v16bf ag2 = *((const v16bf*)(g_lds + 2 * 512 + lane * 16));
    v16bf ag3 = *((const v16bf*)(g_lds + 3 * 512 + lane * 16));
    for (int i = 0; i < 8; ++i) {
        int nt = w + 8 * i;                  // 64 N-tiles across 8 waves
        v16bf bm = *((const v16bf*)(upp + (size_t)nt * 512 + lane * 16));
        v8f au[4];
        au[0] = __builtin_amdgcn_wmma_f32_16x16x32_bf16(false, ag0, false, bm, (short)0, zero, false, false);
        au[1] = __builtin_amdgcn_wmma_f32_16x16x32_bf16(false, ag1, false, bm, (short)0, zero, false, false);
        au[2] = __builtin_amdgcn_wmma_f32_16x16x32_bf16(false, ag2, false, bm, (short)0, zero, false, false);
        au[3] = __builtin_amdgcn_wmma_f32_16x16x32_bf16(false, ag3, false, bm, (short)0, zero, false, false);
        int h = nt * 16 + (lane & 15);
        float wbv = wbbuf[batch * H_DIM + h];
        #pragma unroll
        for (int mt = 0; mt < 4; ++mt)
            #pragma unroll
            for (int v = 0; v < 8; ++v) {
                int rl = mt * 16 + v + (lane >> 4) * 8;
                size_t idx = ((size_t)(rowbase + rl)) * H_DIM + h;
                out[idx] = x[idx] + SCALE_F * (au[mt][v] + wbv) * maskv[rl];
            }
    }
}

// ---------------------------------------------------------------------------

extern "C" void kernel_launch(void* const* d_in, const int* in_sizes, int n_in,
                              void* d_out, int out_size, void* d_ws, size_t ws_size,
                              hipStream_t stream) {
    const float* x  = (const float*)d_in[0];
    const float* w1 = (const float*)d_in[1];
    const float* b1 = (const float*)d_in[2];
    const float* w2 = (const float*)d_in[3];
    const float* b2 = (const float*)d_in[4];
    const float* dw = (const float*)d_in[5];
    const float* db = (const float*)d_in[6];
    const float* uw = (const float*)d_in[7];
    const float* ub = (const float*)d_in[8];
    const float* sw = (const float*)d_in[9];
    const float* sb = (const float*)d_in[10];
    float* out = (float*)d_out;

    char* ws = (char*)d_ws;
    unsigned short* w1p = (unsigned short*)ws;                        // 1 MB
    unsigned short* dwp = (unsigned short*)(ws + (1u << 20));         // 64 KB
    unsigned short* upp = (unsigned short*)(ws + (1u << 20) + (64u << 10)); // 64 KB
    float* awbuf = (float*)(ws + (1u << 20) + (128u << 10));          // 128 B
    float* wbbuf = (float*)(ws + (1u << 20) + (128u << 10) + 128);    // 32 KB

    // allow >64KB dynamic LDS (gfx1250: up to 320KB per workgroup)
    (void)hipFuncSetAttribute((const void*)crft_main,
                              hipFuncAttributeMaxDynamicSharedMemorySize, SMEM_TOTAL);

    crft_pack_w1<<<2048, 256, 0, stream>>>(w1, w1p);
    crft_pack_dw<<<128, 256, 0, stream>>>(dw, dwp);
    crft_pack_up<<<128, 256, 0, stream>>>(uw, upp);
    crft_adapter<<<8, 256, 0, stream>>>(x, sw, sb, ub, awbuf, wbbuf);

    crft_main<<<NROWS / MB, 256, SMEM_TOTAL, stream>>>(
        x, b1, w2, b2, db, w1p, dwp, upp, awbuf, wbbuf, out);
}
